// WaveNetGenerator_14723147891161
// MI455X (gfx1250) — matrix operand order repaired and checked by reading, not compiled, not run on hardware
//
#include <hip/hip_runtime.h>

// ---------------------------------------------------------------------------
// Types
// ---------------------------------------------------------------------------
typedef _Float16 half_t;
typedef __attribute__((ext_vector_type(16))) _Float16      v16h;
typedef __attribute__((ext_vector_type(8)))  float         v8f;
typedef __attribute__((ext_vector_type(4)))  unsigned int  u32x4;
typedef __attribute__((ext_vector_type(8)))  int           i32x8;
typedef __attribute__((ext_vector_type(4)))  int           i32x4;

#define BB    2
#define CC    64
#define LSONG 1024
#define RR    64
#define SS    256
#define NL    8
#define PP    256
#define EPSF  1e-5f

// Packed-weight layout (units: halves). Per-layer block = [dil | res | skip].
#define DIL_H    16384          // 8 mt * 4 kc * 512
#define RES_OFF  16384          // 4 mt * 2 kc * 512 = 4096
#define SKIP_OFF 20480          // 16 mt * 2 kc * 512 = 16384
#define LAYER_H  36864          // = 73728 bytes
#define POUT_OFF 294912         // 16 mt * 8 kc * 512 = 65536
#define PEND_OFF 360448         // 4 mt * 8 kc * 512 = 16384
#define PEMB_OFF 376832         // 4 mt * 8 kc * 512 = 16384
#define PACK_H   393216

// Generator dynamic-LDS layout (bytes)
#define SM_WBUF   0             // 2 * 73728
#define SM_HIST   147456        // 65280 halves = 130560 B  (activation rings)
#define SM_Z32    278016        // 2*128 f32
#define SM_G16    279040        // 2*64  f16
#define SM_SKIP   279296        // 2*256 f32
#define SM_H16    281344        // 2*256 f16
#define SM_H2F    282368        // 2*256 f32
#define SM_H216   284416        // 2*256 f16
#define SM_TOTAL  285696

#if __has_builtin(__builtin_amdgcn_tensor_load_to_lds) && __has_builtin(__builtin_amdgcn_s_wait_tensorcnt)
#define GEN_USE_TDM 1
#else
#define GEN_USE_TDM 0
#endif

// D = A(16x32 f16) x B(32x16 f16) + C(f32)   -- codegen-confirmed builtin
__device__ __forceinline__ v8f wmma16(v16h a, v16h b, v8f c) {
  return __builtin_amdgcn_wmma_f32_16x16x32_f16(false, a, false, b, (short)0, c,
                                                false, false);
}

// ---------------------------------------------------------------------------
// Downsample: y[b,o,l] = sum_c w[o,c,0]x[b,c,2l] + w[o,c,1]x[b,c,2l+1]
// then BN over (b,l) + ReLU.  One block per output channel o.
// ---------------------------------------------------------------------------
__global__ void __launch_bounds__(256)
down_kernel(const float* __restrict__ x, const float* __restrict__ w,
            const float* __restrict__ gamma, const float* __restrict__ beta,
            float* __restrict__ y, int Lin) {
  __shared__ float ws[128];
  __shared__ float redS[256], redQ[256];
  const int o = blockIdx.x, tid = threadIdx.x;
  const int Lout = Lin >> 1, n = BB * Lout;
  if (tid < 128) ws[tid] = w[o * 128 + tid];   // ws[c*2+k]
  __syncthreads();
  float vals[4];
  int   cnt = 0;
  float sum = 0.f, sq = 0.f;
  for (int idx = tid; idx < n; idx += 256) {
    const int b = idx / Lout, ll = idx % Lout;
    const float* xp = x + (size_t)b * CC * Lin + 2 * ll;
    float acc = 0.f;
#pragma unroll 8
    for (int c = 0; c < CC; ++c)
      acc += ws[c * 2] * xp[(size_t)c * Lin] + ws[c * 2 + 1] * xp[(size_t)c * Lin + 1];
    vals[cnt++] = acc; sum += acc; sq += acc * acc;
  }
  redS[tid] = sum; redQ[tid] = sq; __syncthreads();
  for (int s = 128; s > 0; s >>= 1) {
    if (tid < s) { redS[tid] += redS[tid + s]; redQ[tid] += redQ[tid + s]; }
    __syncthreads();
  }
  const float m   = redS[0] / (float)n;
  const float var = redQ[0] / (float)n - m * m;
  const float sc  = gamma[o] * rsqrtf(var + EPSF);
  const float sh  = beta[o] - m * sc;
  cnt = 0;
  for (int idx = tid; idx < n; idx += 256) {
    const int b = idx / Lout, ll = idx % Lout;
    float t = vals[cnt++] * sc + sh;
    y[((size_t)b * CC + o) * Lout + ll] = t > 0.f ? t : 0.f;
  }
}

// ---------------------------------------------------------------------------
// Upsample: y[b,o,2l+k] = sum_i w[i,o,k] x[b,i,l]; BN over (b,2l) + ReLU.
// ---------------------------------------------------------------------------
__global__ void __launch_bounds__(256)
up_kernel(const float* __restrict__ x, const float* __restrict__ w,
          const float* __restrict__ gamma, const float* __restrict__ beta,
          float* __restrict__ y, int Lin) {
  __shared__ float ws[128];                    // ws[i*2+k] = w[(i*64+o)*2+k]
  __shared__ float redS[256], redQ[256];
  const int o = blockIdx.x, tid = threadIdx.x;
  const int Lout = 2 * Lin, n = BB * Lout;
  if (tid < 64) { ws[tid * 2] = w[(tid * 64 + o) * 2]; ws[tid * 2 + 1] = w[(tid * 64 + o) * 2 + 1]; }
  __syncthreads();
  float vals[8];
  int   cnt = 0;
  float sum = 0.f, sq = 0.f;
  for (int idx = tid; idx < n; idx += 256) {
    const int b = idx / Lout, p = idx % Lout, l = p >> 1, k = p & 1;
    const float* xp = x + (size_t)b * CC * Lin + l;
    float acc = 0.f;
#pragma unroll 8
    for (int i = 0; i < CC; ++i) acc += ws[i * 2 + k] * xp[(size_t)i * Lin];
    vals[cnt++] = acc; sum += acc; sq += acc * acc;
  }
  redS[tid] = sum; redQ[tid] = sq; __syncthreads();
  for (int s = 128; s > 0; s >>= 1) {
    if (tid < s) { redS[tid] += redS[tid + s]; redQ[tid] += redQ[tid + s]; }
    __syncthreads();
  }
  const float m   = redS[0] / (float)n;
  const float var = redQ[0] / (float)n - m * m;
  const float sc  = gamma[o] * rsqrtf(var + EPSF);
  const float sh  = beta[o] - m * sc;
  cnt = 0;
  for (int idx = tid; idx < n; idx += 256) {
    const int b = idx / Lout, p = idx % Lout;
    float t = vals[cnt++] * sc + sh;
    y[((size_t)b * CC + o) * Lout + p] = t > 0.f ? t : 0.f;
  }
}

// ---------------------------------------------------------------------------
// cond GEMM (WMMA): cond[b,t,o] = sum_c w_cond[o,c] * xds[b,c,t]
// M=1024 (o), N=512 (b*256+t), K=64.  One wave per 16x16 tile.
// ---------------------------------------------------------------------------
__global__ void __launch_bounds__(256)
cond_kernel(const float* __restrict__ w_cond, const float* __restrict__ xds,
            float* __restrict__ cond) {
  const int tid  = threadIdx.x, lane = tid & 31, wave = tid >> 5;
  const int wid  = blockIdx.x * 8 + wave;       // 0..2047
  const int mt   = wid & 63, nt = wid >> 6;     // 64 M-tiles, 32 N-tiles
  const int m    = mt * 16 + (lane & 15);
  const int n    = nt * 16 + (lane & 15);
  const int bix  = n >> 8, t = n & 255;
  const int kbA  = (lane & 16) ? 8 : 0;
  const int kbB  = (lane & 16) ? 16 : 0;
  v8f d = {};
#pragma unroll
  for (int kc = 0; kc < 2; ++kc) {
    v16h af, bf;
#pragma unroll
    for (int h = 0; h < 16; ++h) {
      const int kA = kc * 32 + kbB + h;
      af[h] = (half_t)w_cond[m * CC + kA];
      const int kB = kc * 32 + kbB + h;
      bf[h] = (half_t)xds[((size_t)bix * CC + kB) * 256 + t];
    }
    d = wmma16(af, bf, d);
  }
  const int row0 = mt * 16 + ((lane & 16) ? 8 : 0);
  float* op = cond + ((size_t)bix * 256 + t) * 1024 + row0;
#pragma unroll
  for (int r = 0; r < 8; ++r) op[r] = d[r];
}

// ---------------------------------------------------------------------------
// w_emb_end = w_embed(64x64) @ w_end(64x256)
// ---------------------------------------------------------------------------
__global__ void __launch_bounds__(256)
embend_kernel(const float* __restrict__ w_embed, const float* __restrict__ w_end,
              float* __restrict__ out) {
  const int idx = blockIdx.x * 256 + threadIdx.x;
  if (idx >= RR * SS) return;
  const int r = idx >> 8, t = idx & 255;
  float acc = 0.f;
#pragma unroll 8
  for (int c = 0; c < CC; ++c) acc += w_embed[r * CC + c] * w_end[c * SS + t];
  out[r * SS + t] = acc;
}

// ---------------------------------------------------------------------------
// Pack all generator weights to f16 WMMA A-fragment order.
// A-frag(mt,kc): lane L row m0+(L&15); half h -> K = kc*32 + (L&16?16:0) + h
// ---------------------------------------------------------------------------
__global__ void __launch_bounds__(256)
pack_kernel(const float* __restrict__ w_dil, const float* __restrict__ w_res,
            const float* __restrict__ w_skip, const float* __restrict__ w_out,
            const float* __restrict__ w_end, const float* __restrict__ w_emb_end,
            half_t* __restrict__ packed) {
  const int idx = blockIdx.x * 256 + threadIdx.x;
  if (idx >= PACK_H) return;
  int f, q, kcw, l = 0, region;
  int rr;
  if (idx < POUT_OFF) {
    l = idx / LAYER_H; rr = idx % LAYER_H;
    if (rr < RES_OFF)        { region = 0; rr -= 0;        kcw = 4; }
    else if (rr < SKIP_OFF)  { region = 1; rr -= RES_OFF;  kcw = 2; }
    else                     { region = 2; rr -= SKIP_OFF; kcw = 2; }
  } else if (idx < PEND_OFF) { region = 3; rr = idx - POUT_OFF; kcw = 8; }
  else if (idx < PEMB_OFF)   { region = 4; rr = idx - PEND_OFF; kcw = 8; }
  else                       { region = 5; rr = idx - PEMB_OFF; kcw = 8; }
  f = rr / 512; q = rr % 512;
  const int lane = q >> 4, h = q & 15;
  const int mt = f / kcw, kc = f % kcw;
  const int m  = mt * 16 + (lane & 15);
  const int k  = kc * 32 + ((lane & 16) ? 16 : 0) + h;
  float v;
  switch (region) {
    case 0:  // dil: K<64 -> tap0 on x[a-d], else tap1 on x[a]
      v = (k < 64) ? w_dil[(((size_t)l * 128 + m) * 64 + k) * 2 + 0]
                   : w_dil[(((size_t)l * 128 + m) * 64 + (k - 64)) * 2 + 1];
      break;
    case 1: v = w_res[((size_t)l * 64 + m) * 64 + k];   break;
    case 2: v = w_skip[((size_t)l * 256 + m) * 64 + k]; break;
    case 3: v = w_out[m * 256 + k];                     break;
    case 4: v = w_end[m * 256 + k];                     break;
    default: v = w_emb_end[m * 256 + k];                break;
  }
  packed[idx] = (half_t)v;
}

__global__ void zero_kernel(float* p, int n) {
  int i = blockIdx.x * 256 + threadIdx.x;
  if (i < n) p[i] = 0.f;
}

// ---------------------------------------------------------------------------
// TDM: stage one 73728-byte layer block (9 rows x 4096 halves) into LDS.
// This toolchain exposes the 6-arg builtin:
//   (uint32x4 g0, int32x8 g1, int32x4 g2, int32x4 g3, int32x8 g4, i32 cpol)
// groups 2/3/4 are only needed for >2D tensors -> zeros.
// ---------------------------------------------------------------------------
__device__ __forceinline__ void stage_layer(const half_t* gsrc, half_t* dstLds,
                                            int lane) {
#if GEN_USE_TDM
  const unsigned long long ga = (unsigned long long)(uintptr_t)gsrc;
  u32x4 g0; i32x8 g1;
  i32x4 g2 = {0, 0, 0, 0}, g3 = {0, 0, 0, 0};
  i32x8 g4 = {0, 0, 0, 0, 0, 0, 0, 0};
  g0[0] = 1u;                                          // count=1, user D#
  g0[1] = (unsigned)(uintptr_t)dstLds;                 // lds_addr (bytes)
  g0[2] = (unsigned)(ga & 0xffffffffull);              // global_addr[31:0]
  g0[3] = (unsigned)((ga >> 32) & 0x1ffffffull) | 0x80000000u; // addr[56:32]|type=2
  g1[0] = (int)(1u << 16);                             // data_size=2B
  g1[1] = (int)(4096u << 16);                          // tensor_dim0 lo16
  g1[2] = (int)(9u << 16);                             // dim0 hi16 | tensor_dim1 lo16
  g1[3] = (int)(4096u << 16);                          // dim1 hi16 | tile_dim0
  g1[4] = (int)9;                                      // tile_dim1 | tile_dim2=0
  g1[5] = (int)4096;                                   // tensor_dim0_stride lo32
  g1[6] = 0; g1[7] = 0;
  __builtin_amdgcn_tensor_load_to_lds(g0, g1, g2, g3, g4, 0);
#else
  for (int q = lane; q < LAYER_H / 16; q += 32)
    ((v16h*)dstLds)[q] = ((const v16h*)gsrc)[q];
#endif
}

__device__ __forceinline__ void wait_stage() {
#if GEN_USE_TDM
  __builtin_amdgcn_s_wait_tensorcnt(0);
#endif
}

// ---------------------------------------------------------------------------
// Autoregressive generator: 1 workgroup, 8 waves, 255 sequential steps.
// Both batch elements occupy B-matrix columns 0 and 1.
// ---------------------------------------------------------------------------
__global__ void __launch_bounds__(256)
gen_kernel(const half_t* __restrict__ packed, const float* __restrict__ cond,
           float* __restrict__ gen_out) {
  extern __shared__ char smem[];
  half_t* wbuf   = (half_t*)(smem + SM_WBUF);
  half_t* hist   = (half_t*)(smem + SM_HIST);
  float*  z32    = (float*)(smem + SM_Z32);
  half_t* g16    = (half_t*)(smem + SM_G16);
  float*  skip32 = (float*)(smem + SM_SKIP);
  half_t* h16    = (half_t*)(smem + SM_H16);
  float*  h2f    = (float*)(smem + SM_H2F);
  half_t* h216   = (half_t*)(smem + SM_H216);

  const int tid  = threadIdx.x;
  const int lane = tid & 31;
  const int wave = tid >> 5;
  const int col  = lane & 15;
  const int rb   = (lane & 16) ? 8 : 0;   // D-frag row offset
  const int kbB  = (lane & 16) ? 16 : 0;  // B-frag K offset
  const int bcl  = (col < 2) ? col : 0;   // clamped batch (junk cols harmless)
  const bool vld = (col < 2);

  // zero activation rings (65280 halves)
  for (int k = tid; k < 65280 / 2; k += 256) ((int*)hist)[k] = 0;
  __syncthreads();
  if (wave == 0) stage_layer(packed, wbuf, lane);      // layer 0 -> buf 0

  const half_t* pout = packed + POUT_OFF;
  const half_t* pend = packed + PEND_OFF;
  const half_t* pemb = packed + PEMB_OFF;

  for (int i = 0; i < 255; ++i) {
    const int a = 255 + i;
    v8f sk0 = {}, sk1 = {};

    for (int l = 0; l < NL; ++l) {
      if (wave == 0) wait_stage();      // layer-l weights present in LDS
      __syncthreads();
      if (wave == 0)                    // prefetch next layer's block
        stage_layer(packed + (size_t)((l + 1) & 7) * LAYER_H,
                    wbuf + ((l + 1) & 1) * LAYER_H, lane);
      if (l < 6) {                      // warm L2->near for output-stage weights
        __builtin_prefetch((const char*)pout + ((size_t)(l * 256 + tid)) * 128, 0, 1);
      }
      const half_t* W    = wbuf + (l & 1) * LAYER_H;
      const int d        = 1 << l;
      const int ringOff  = 256 * ((1 << l) - 1);
      const int mask     = (2 << l) - 1;
      const int sNew     = a & mask;
      const int sOld     = (a - d) & mask;

      // ---- z = Wdil0 x[a-d] + Wdil1 x[a]  (128 rows; one M-tile per wave)
      v8f zc = {};
#pragma unroll
      for (int kc = 0; kc < 4; ++kc) {
        v16h af = *(const v16h*)(W + (wave * 4 + kc) * 512 + lane * 16);
        const int k0 = kc * 32 + kbB;
        const int slot = (k0 >= 64) ? sNew : sOld;
        v16h bf = *(const v16h*)(hist + ringOff + slot * 128 + bcl * 64 + (k0 & 63));
        zc = wmma16(af, bf, zc);
      }
      if (vld) {
        const float* cp = cond + ((size_t)bcl * 256 + i) * 1024 + l * 128 + wave * 16 + rb;
        float* zp = z32 + bcl * 128 + wave * 16 + rb;
#pragma unroll
        for (int r = 0; r < 8; ++r) zp[r] = zc[r] + cp[r];
      }
      __syncthreads();

      // ---- gating: g = tanh(z[:R]) * sigmoid(z[R:])
      if (tid < 128) {
        const int b = tid >> 6, ch = tid & 63;
        const float z0 = z32[b * 128 + ch];
        const float z1 = z32[b * 128 + 64 + ch];
        const float g  = tanhf(z0) * (1.0f / (1.0f + __expf(-z1)));
        g16[b * 64 + ch] = (half_t)g;
      }
      __syncthreads();

      // ---- residual: x_{l+1}[a] = x_l[a] + Wres g   (waves 0..3)
      if (wave < 4 && l < 7) {
        v8f rc = {};
#pragma unroll
        for (int kc = 0; kc < 2; ++kc) {
          v16h af = *(const v16h*)(W + RES_OFF + (wave * 2 + kc) * 512 + lane * 16);
          v16h bf = *(const v16h*)(g16 + bcl * 64 + kc * 32 + kbB);
          rc = wmma16(af, bf, rc);
        }
        if (vld) {
          const int row0     = wave * 16 + rb;
          const int ringOff1 = 256 * ((2 << l) - 1);
          const int sN1      = a & ((4 << l) - 1);
          half_t*       xp = hist + ringOff1 + sN1 * 128 + bcl * 64 + row0;
          const half_t* xc = hist + ringOff + sNew * 128 + bcl * 64 + row0;
#pragma unroll
          for (int r = 0; r < 8; ++r) xp[r] = (half_t)((float)xc[r] + rc[r]);
        }
      }
      // ---- skip accumulate (register-resident across layers; all waves, 2 tiles)
#pragma unroll
      for (int kc = 0; kc < 2; ++kc) {
        v16h bf  = *(const v16h*)(g16 + bcl * 64 + kc * 32 + kbB);
        v16h af0 = *(const v16h*)(W + SKIP_OFF + (wave * 2 + kc) * 512 + lane * 16);
        v16h af1 = *(const v16h*)(W + SKIP_OFF + ((wave + 8) * 2 + kc) * 512 + lane * 16);
        sk0 = wmma16(af0, bf, sk0);
        sk1 = wmma16(af1, bf, sk1);
      }
      __syncthreads();
    }

    // ---- output head: h = relu(skip); h2 = relu(Wout h); y = Wend h2
    if (vld) {
#pragma unroll
      for (int r = 0; r < 8; ++r) {
        skip32[bcl * 256 + wave * 16 + rb + r]        = sk0[r];
        skip32[bcl * 256 + (wave + 8) * 16 + rb + r]  = sk1[r];
      }
    }
    __syncthreads();
    for (int k = tid; k < 512; k += 256) {
      float v = skip32[k]; h16[k] = (half_t)(v > 0.f ? v : 0.f);
    }
    __syncthreads();
#pragma unroll
    for (int t2 = 0; t2 < 2; ++t2) {
      const int mt = wave + t2 * 8;
      v8f oc = {};
#pragma unroll
      for (int kc = 0; kc < 8; ++kc) {
        v16h af = *(const v16h*)(pout + (mt * 8 + kc) * 512 + lane * 16);
        v16h bf = *(const v16h*)(h16 + bcl * 256 + kc * 32 + kbB);
        oc = wmma16(af, bf, oc);
      }
      if (vld) {
#pragma unroll
        for (int r = 0; r < 8; ++r) h2f[bcl * 256 + mt * 16 + rb + r] = oc[r];
      }
    }
    __syncthreads();
    for (int k = tid; k < 512; k += 256) {
      float v = h2f[k]; h216[k] = (half_t)(v > 0.f ? v : 0.f);
    }
    __syncthreads();
    if (wave < 4) {                       // y -> gen_out[:, :, i+1]
      v8f yc = {};
#pragma unroll
      for (int kc = 0; kc < 8; ++kc) {
        v16h af = *(const v16h*)(pend + (wave * 8 + kc) * 512 + lane * 16);
        v16h bf = *(const v16h*)(h216 + bcl * 256 + kc * 32 + kbB);
        yc = wmma16(af, bf, yc);
      }
      if (vld) {
        const int row0 = wave * 16 + rb;
#pragma unroll
        for (int r = 0; r < 8; ++r)
          gen_out[((size_t)bcl * CC + row0 + r) * 256 + (i + 1)] = yc[r];
      }
    } else {                              // x0[a+1] = (w_embed@w_end) h2 -> ring 0
      const int mt = wave - 4;
      v8f xc = {};
#pragma unroll
      for (int kc = 0; kc < 8; ++kc) {
        v16h af = *(const v16h*)(pemb + (mt * 8 + kc) * 512 + lane * 16);
        v16h bf = *(const v16h*)(h216 + bcl * 256 + kc * 32 + kbB);
        xc = wmma16(af, bf, xc);
      }
      if (vld) {
        const int row0 = mt * 16 + rb;
        half_t* xp = hist + ((a + 1) & 1) * 128 + bcl * 64 + row0;
#pragma unroll
        for (int r = 0; r < 8; ++r) xp[r] = (half_t)xc[r];
      }
    }
    __syncthreads();
  }
}

// ---------------------------------------------------------------------------
// Host orchestration
// ---------------------------------------------------------------------------
extern "C" void kernel_launch(void* const* d_in, const int* in_sizes, int n_in,
                              void* d_out, int out_size, void* d_ws, size_t ws_size,
                              hipStream_t stream) {
  const float* song    = (const float*)d_in[0];
  const float* ds_w    = (const float*)d_in[1];
  const float* ds_g    = (const float*)d_in[2];
  const float* ds_b    = (const float*)d_in[3];
  const float* us_w    = (const float*)d_in[4];
  const float* us_g    = (const float*)d_in[5];
  const float* us_b    = (const float*)d_in[6];
  const float* w_embed = (const float*)d_in[7];
  const float* w_cond  = (const float*)d_in[8];
  const float* w_dil   = (const float*)d_in[9];
  const float* w_res   = (const float*)d_in[10];
  const float* w_skip  = (const float*)d_in[11];
  const float* w_out   = (const float*)d_in[12];
  const float* w_end   = (const float*)d_in[13];
  float* out = (float*)d_out;

  char* ws = (char*)d_ws;
  size_t off = 0;
  auto alloc = [&](size_t bytes) { char* p = ws + off; off = (off + bytes + 255) & ~(size_t)255; return p; };
  float*  xds1    = (float*)alloc(2 * 64 * 512 * 4);
  float*  xds2    = (float*)alloc(2 * 64 * 256 * 4);
  float*  condb   = (float*)alloc((size_t)2 * 256 * 1024 * 4);
  float*  wembend = (float*)alloc(64 * 256 * 4);
  half_t* packed  = (half_t*)alloc((size_t)PACK_H * 2);
  float*  gen     = (float*)alloc(2 * 64 * 256 * 4);
  float*  up1     = (float*)alloc(2 * 64 * 512 * 4);

  // downsample x2
  down_kernel<<<64, 256, 0, stream>>>(song, ds_w, ds_g, ds_b, xds1, 1024);
  down_kernel<<<64, 256, 0, stream>>>(xds1, ds_w + 64 * 64 * 2, ds_g + 64, ds_b + 64, xds2, 512);
  // conditioning GEMM (WMMA)
  cond_kernel<<<256, 256, 0, stream>>>(w_cond, xds2, condb);
  // combined embed@end + packed f16 weights
  embend_kernel<<<64, 256, 0, stream>>>(w_embed, w_end, wembend);
  pack_kernel<<<PACK_H / 256, 256, 0, stream>>>(w_dil, w_res, w_skip, w_out, w_end,
                                                wembend, packed);
  zero_kernel<<<(2 * 64 * 256 + 255) / 256, 256, 0, stream>>>(gen, 2 * 64 * 256);
  // sequential autoregressive generation (single workgroup)
  gen_kernel<<<1, 256, SM_TOTAL, stream>>>(packed, condb, gen);
  // upsample x2
  up_kernel<<<64, 256, 0, stream>>>(gen, us_w, us_g, us_b, up1, 256);
  up_kernel<<<64, 256, 0, stream>>>(up1, us_w + 64 * 64 * 2, us_g + 64, us_b + 64, out, 512);
}
